// LatentMultipleInterShooting_52140902974147
// MI455X (gfx1250) — matrix operand order repaired and checked by reading, not compiled, not run on hardware
//
#include <hip/hip_runtime.h>
#include <hip/hip_bf16.h>
#include <stdint.h>

// ---------------------------------------------------------------------------
// Problem constants
// ---------------------------------------------------------------------------
#define NB   131073           // tokens per sequence slot (L_SIG)
#define DLS  132              // LDS row stride (floats) for 128-wide f32 mats
#define XS   132              // xbuf stride (floats)
#define HS   388              // hbuf stride (floats)  (>= 384, bank-skewed)

// workspace layout (float offsets)
#define MT_OFF    0           // Mt               128x128
#define R_OFF     16384       // Mt^r, r=0..15    16 x 128x128
#define M256_OFF  278528      // Mt^256           128x128
#define S_OFF     294912      // S_i = V*Mt^256i  9 x 129x128
#define WB_BYTE_OFF 1835008   // bf16 swizzled weights (float off 458752 * 4)

typedef __attribute__((ext_vector_type(16))) __bf16 v16bf;
typedef __attribute__((ext_vector_type(8)))  float  v8f;

#if __has_builtin(__builtin_amdgcn_wmma_f32_16x16x4_f32)
#define HAVE_WMMA_F32X4 1
typedef __attribute__((ext_vector_type(2))) float v2f;
#endif

// ---------------------------------------------------------------------------
// Async global<->LDS movers (CDNA5, tracked by ASYNCcnt). Data bypasses VGPRs.
// ---------------------------------------------------------------------------
#define USE_ASYNC 1

#if USE_ASYNC
__device__ __forceinline__ uint32_t lds_off(const void* p) {
  // low 32 bits of a generic pointer into the LDS aperture = LDS byte offset
  return (uint32_t)(uintptr_t)p;
}
__device__ __forceinline__ void async_ld_b128(const void* lds, const void* g) {
  asm volatile("global_load_async_to_lds_b128 %0, %1, off"
               :: "v"(lds_off(lds)), "v"(g) : "memory");
}
__device__ __forceinline__ void async_st_b128(const void* g, const void* lds) {
  asm volatile("global_store_async_from_lds_b128 %0, %1, off"
               :: "v"(g), "v"(lds_off(lds)) : "memory");
}
__device__ __forceinline__ void wait_async0() {
  asm volatile("s_wait_asynccnt 0x0" ::: "memory");
}
#endif

// ---------------------------------------------------------------------------
// K0: swizzle f32 weights into bf16 WMMA B-fragment order.
// ---------------------------------------------------------------------------
__device__ __forceinline__ void swz_one(const float* __restrict__ src,
                                        __hip_bfloat16* __restrict__ dst,
                                        int K, int N, int e) {
  int KC = K >> 5;
  int f = e >> 9;             // 512 elements per fragment
  int r = e & 511;
  int lane = r >> 4;
  int j = r & 15;
  int nt = f / KC;
  int kc = f - nt * KC;
  int k = kc * 32 + ((lane & 16) ? 16 : 0) + j;
  int n = nt * 16 + (lane & 15);
  dst[e] = __float2bfloat16(src[k * N + n]);
}

__global__ void __launch_bounds__(256)
k0_swz(const float* qkv_w, const float* out_w, const float* ff1_w,
       const float* ff2_w, const float* dec_w, __hip_bfloat16* wb) {
  int e = blockIdx.x * 256 + threadIdx.x;
  const int PER_L = 131072;
  if (e < 5 * PER_L) {
    int l = e / PER_L, o = e - l * PER_L;
    __hip_bfloat16* dl = wb + l * PER_L;
    if (o < 49152)      swz_one(qkv_w + l * 49152, dl,          128, 384, o);
    else if (o < 65536) swz_one(out_w + l * 16384, dl + 49152,  128, 128, o - 49152);
    else if (o < 98304) swz_one(ff1_w + l * 32768, dl + 65536,  128, 256, o - 65536);
    else                swz_one(ff2_w + l * 32768, dl + 98304,  256, 128, o - 98304);
  } else if (e < 5 * PER_L + 8192) {
    swz_one(dec_w, wb + 5 * PER_L, 128, 64, e - 5 * PER_L);
  }
}

// ---------------------------------------------------------------------------
// C[M x 128] (+)= scale * A[M x 128] @ B[128 x 128]   (LDS, stride DLS)
// Rows 0..127 on the f32 WMMA path (16x16x4, 32 A-fragments hoisted into
// registers); row 128 (M==129) via a VALU tail.  Used by K1 and K2 — these
// phases must stay f32 (Mt = I + O(1e-7); bf16 would round Mt to identity).
// ---------------------------------------------------------------------------
__device__ void wg_gemm_f32(const float* A, const float* B, float* C,
                            int M, float scale, bool accum) {
  __syncthreads();
#if HAVE_WMMA_F32X4
  const int lane = threadIdx.x & 31;
  const int wave = threadIdx.x >> 5;
  const int m = wave * 16 + (lane & 15);
  const int koff = (lane & 16) ? 2 : 0;
  const int n15 = lane & 15;
  const int row0 = wave * 16 + ((lane & 16) ? 8 : 0);
  v2f areg[32];
#pragma unroll
  for (int kc = 0; kc < 32; ++kc) {
    int k0 = kc * 4 + koff;
    v2f a; a[0] = A[m * DLS + k0]; a[1] = A[m * DLS + k0 + 1];
    areg[kc] = a;
  }
  for (int nt = 0; nt < 8; ++nt) {
    int n = nt * 16 + n15;
    v8f acc;
#pragma unroll
    for (int v = 0; v < 8; ++v) acc[v] = 0.f;
#pragma unroll
    for (int kc = 0; kc < 32; ++kc) {
      int k0 = kc * 4 + koff;
      v2f b; b[0] = B[k0 * DLS + n]; b[1] = B[(k0 + 1) * DLS + n];
      acc = __builtin_amdgcn_wmma_f32_16x16x4_f32(false, areg[kc], false, b,
                                                  (short)0, acc, false, false);
    }
#pragma unroll
    for (int v = 0; v < 8; ++v) {
      float out = scale * acc[v];
      if (accum) out += C[(row0 + v) * DLS + n];
      C[(row0 + v) * DLS + n] = out;
    }
  }
  if (M == 129 && threadIdx.x < 128) {     // row 128 tail
    int c = threadIdx.x;
    float s = 0.f;
#pragma unroll 4
    for (int k = 0; k < 128; ++k) s = fmaf(A[128 * DLS + k], B[k * DLS + c], s);
    float out = scale * s;
    if (accum) out += C[128 * DLS + c];
    C[128 * DLS + c] = out;
  }
#else
  for (int e = threadIdx.x; e < M * 128; e += 256) {
    int r = e >> 7, c = e & 127;
    const float* ar = A + r * DLS;
    float s = 0.f;
#pragma unroll 4
    for (int k = 0; k < 128; ++k) s = fmaf(ar[k], B[k * DLS + c], s);
    float out = scale * s;
    if (accum) out += C[r * DLS + c];
    C[r * DLS + c] = out;
  }
#endif
  __syncthreads();
}

__device__ void wg_store_g(float* g, const float* src, int M) {
  __syncthreads();
  for (int e = threadIdx.x; e < M * 128; e += 256)
    g[e] = src[(e >> 7) * DLS + (e & 127)];
  __syncthreads();
}

__device__ void wg_copy(float* dst, const float* src, int M) {
  __syncthreads();
  for (int e = threadIdx.x; e < M * 128; e += 256)
    dst[(e >> 7) * DLS + (e & 127)] = src[(e >> 7) * DLS + (e & 127)];
  __syncthreads();
}

// ---------------------------------------------------------------------------
// K1 (1 workgroup): Mt = (I + dtA + dt^2/2 A^2 + dt^3/6 A^3 + dt^4/24 A^4)^T
// then power table Mt^r (r=0..15), Mt^256, and checkpoints S_i = V*Mt^256i.
// ---------------------------------------------------------------------------
__global__ void __launch_bounds__(256)
k1_mats(const float* A, const float* t, const float* V, float* ws) {
  extern __shared__ float sm[];
  float* b0 = sm;                    // 129 x DLS each
  float* b1 = sm + 129 * DLS;
  float* b2 = sm + 2 * 129 * DLS;
  const int tid = threadIdx.x;
  const float dt = t[1] - t[0];
  const float c2 = dt * dt * 0.5f;
  const float c3 = c2 * dt * (1.f / 3.f);
  const float c4 = c3 * dt * 0.25f;

  for (int e = tid; e < 128 * 128; e += 256) {   // b0 = A^T
    int r = e >> 7, c = e & 127;
    b0[r * DLS + c] = A[c * 128 + r];
  }
  wg_gemm_f32(b0, b0, b1, 128, 1.f, false);      // b1 = At^2
  wg_gemm_f32(b1, b0, b2, 128, 1.f, false);      // b2 = At^3
  __syncthreads();
  for (int e = tid; e < 128 * 128; e += 256) {   // b1 = I + dt At + c2 At^2 + c3 At^3
    int r = e >> 7, c = e & 127;
    float v = (r == c ? 1.f : 0.f) + dt * b0[r * DLS + c]
            + c2 * b1[r * DLS + c] + c3 * b2[r * DLS + c];
    b1[r * DLS + c] = v;
  }
  __syncthreads();
  wg_gemm_f32(b2, b0, b1, 128, c4, true);        // b1 += c4 At^4 => Mt

  __syncthreads();
  for (int e = tid; e < 128 * 128; e += 256)     // R[0] = I
    ws[R_OFF + e] = ((e >> 7) == (e & 127)) ? 1.f : 0.f;
  wg_store_g(ws + MT_OFF, b1, 128);
  wg_store_g(ws + R_OFF + 16384, b1, 128);       // R[1] = Mt
  wg_copy(b0, b1, 128);                          // b0 = Mt (fixed operand)

  float* cur = b1; float* nxt = b2;
  for (int r = 2; r <= 15; ++r) {                // R[r] = R[r-1] @ Mt
    wg_gemm_f32(cur, b0, nxt, 128, 1.f, false);
    wg_store_g(ws + R_OFF + r * 16384, nxt, 128);
    float* tmp = cur; cur = nxt; nxt = tmp;
  }
  wg_gemm_f32(cur, b0, nxt, 128, 1.f, false);    // Mt^16
  { float* tmp = cur; cur = nxt; nxt = tmp; }
  for (int q = 0; q < 4; ++q) {                  // ^32 ^64 ^128 ^256
    wg_gemm_f32(cur, cur, nxt, 128, 1.f, false);
    float* tmp = cur; cur = nxt; nxt = tmp;
  }
  wg_store_g(ws + M256_OFF, cur, 128);

  __syncthreads();                               // checkpoints S_0..S_8
  for (int e = tid; e < 129 * 128; e += 256)
    b0[(e >> 7) * DLS + (e & 127)] = V[e];
  wg_store_g(ws + S_OFF, b0, 129);
  float* z = b0; float* zn = nxt;
  for (int i = 1; i <= 8; ++i) {
    wg_gemm_f32(z, cur, zn, 129, 1.f, false);
    wg_store_g(ws + S_OFF + i * 16512, zn, 129);
    float* tmp = z; z = zn; zn = tmp;
  }
}

// ---------------------------------------------------------------------------
// K2 (128 workgroups): segment j covers RK4 steps k = 16j .. 16j+15.
// f32 WMMA steps; scatter uses GLOBAL_STORE_ASYNC_FROM_LDS (overlaps the next
// step's WMMA; wait at iteration end before the buffer becomes a GEMM target).
// ---------------------------------------------------------------------------
__global__ void __launch_bounds__(256)
k2_shoot(const float* ws, float* z2, float* z20, float* z21) {
  extern __shared__ float sm[];
  float* za  = sm;                   // 129 x DLS
  float* zb  = sm + 129 * DLS;       // 129 x DLS
  float* mat = sm + 2 * 129 * DLS;   // 128 x DLS
  const int tid = threadIdx.x;
  const int j = blockIdx.x;
  const int si = j >> 4, rr0 = j & 15;

#if USE_ASYNC
  for (int e = tid; e < 128 * 32; e += 256) {     // mat = R[j%16]
    int r = e >> 5, c = (e & 31) * 4;
    async_ld_b128(mat + r * DLS + c, ws + R_OFF + rr0 * 16384 + r * 128 + c);
  }
  for (int e = tid; e < 129 * 32; e += 256) {     // za = S[j/16]
    int r = e >> 5, c = (e & 31) * 4;
    async_ld_b128(za + r * DLS + c, ws + S_OFF + si * 16512 + r * 128 + c);
  }
  wait_async0();
#else
  for (int e = tid; e < 128 * 128; e += 256)
    mat[(e >> 7) * DLS + (e & 127)] = ws[R_OFF + rr0 * 16384 + e];
  for (int e = tid; e < 129 * 128; e += 256)
    za[(e >> 7) * DLS + (e & 127)] = ws[S_OFF + si * 16512 + e];
#endif
  wg_gemm_f32(za, mat, zb, 129, 1.f, false);      // zb = C_j
  float* z = zb; float* zn = za;
#if USE_ASYNC
  for (int e = tid; e < 128 * 32; e += 256) {     // mat = Mt
    int r = e >> 5, c = (e & 31) * 4;
    async_ld_b128(mat + r * DLS + c, ws + MT_OFF + r * 128 + c);
  }
  wait_async0();
#else
  for (int e = tid; e < 128 * 128; e += 256)
    mat[(e >> 7) * DLS + (e & 127)] = ws[MT_OFF + e];
#endif
  __syncthreads();

  for (int step = 0; step < 16; ++step) {
    int k = j * 16 + step;
#if USE_ASYNC
    if (k < 1024) {                  // z2[0, r*1024+k] = pred_z[k, r+1]
      for (int e = tid; e < 128 * 32; e += 256) {
        int rr = e >> 5, c = (e & 31) * 4;
        const float* ls = z + (rr + 1) * DLS + c;
        size_t b = (size_t)rr * 1024 + k;
        async_st_b128(z2 + b * 128 + c, ls);
        async_st_b128(z20 + b * 128 + c, ls);
      }
    } else {                          // z2[1, r*1024+(k-1024)] = pred_z[k, r]
      int kk = k - 1024;
      for (int e = tid; e < 128 * 32; e += 256) {
        int rr = e >> 5, c = (e & 31) * 4;
        const float* ls = z + rr * DLS + c;
        size_t b = (size_t)rr * 1024 + kk;
        async_st_b128(z2 + ((size_t)NB + b) * 128 + c, ls);
        async_st_b128(z21 + b * 128 + c, ls);
      }
      if (k == 1024) {                // z2[0, 131072] = pred_z[1024, 128]
        for (int e = tid; e < 32; e += 256) {
          int c = e * 4;
          const float* ls = z + 128 * DLS + c;
          async_st_b128(z2 + (size_t)131072 * 128 + c, ls);
          async_st_b128(z20 + (size_t)131072 * 128 + c, ls);
        }
      }
    }
#else
    if (k < 1024) {
      for (int e = tid; e < 128 * 128; e += 256) {
        int rr = e >> 7, c = e & 127;
        float v = z[(rr + 1) * DLS + c];
        size_t b = (size_t)rr * 1024 + k;
        z2[b * 128 + c] = v; z20[b * 128 + c] = v;
      }
    } else {
      int kk = k - 1024;
      for (int e = tid; e < 128 * 128; e += 256) {
        int rr = e >> 7, c = e & 127;
        float v = z[rr * DLS + c];
        size_t b = (size_t)rr * 1024 + kk;
        z2[((size_t)NB + b) * 128 + c] = v; z21[b * 128 + c] = v;
      }
      if (k == 1024) {
        for (int c = tid; c < 128; c += 256) {
          float v = z[128 * DLS + c];
          z2[(size_t)131072 * 128 + c] = v; z20[(size_t)131072 * 128 + c] = v;
        }
      }
    }
#endif
    wg_gemm_f32(z, mat, zn, 129, 1.f, false);  // overlaps in-flight async stores
    float* tmp = z; z = zn; zn = tmp;
#if USE_ASYNC
    wait_async0();                   // old z becomes next GEMM's write target
#endif
  }
  if (j == 127) {                     // z2[1, 131072] = pred_z[2048, 127]
#if USE_ASYNC
    for (int e = tid; e < 32; e += 256) {
      int c = e * 4;
      const float* ls = z + 127 * DLS + c;
      async_st_b128(z2 + ((size_t)NB + 131072) * 128 + c, ls);
      async_st_b128(z21 + (size_t)131072 * 128 + c, ls);
    }
#else
    for (int c = tid; c < 128; c += 256) {
      float v = z[127 * DLS + c];
      z2[((size_t)NB + 131072) * 128 + c] = v;
      z21[(size_t)131072 * 128 + c] = v;
    }
#endif
  }
}

// ---------------------------------------------------------------------------
// K3: fused 5-layer transformer + decoder, one 64-token tile per WG.
// ---------------------------------------------------------------------------
__device__ void mirror128(const float* src, int sstride, int scol,
                          __hip_bfloat16* abufp) {
  __syncthreads();
  uint32_t* ab = (uint32_t*)abufp;
  for (int p = threadIdx.x; p < 8192; p += 256) {   // bf16 pairs
    int idx = p * 2;
    int frag = idx >> 9;          // mtile*4 + kc
    int r = idx & 511;
    int lane = r >> 4;
    int jj = r & 15;              // even
    int mtile = frag >> 2, kc = frag & 3;
    int m = mtile * 16 + (lane & 15);
    int k = kc * 32 + ((lane & 16) ? 16 : 0) + jj;
    const float* q = src + m * sstride + scol + k;
    union { uint32_t u; __hip_bfloat16 h[2]; } cv;
    cv.h[0] = __float2bfloat16(q[0]);
    cv.h[1] = __float2bfloat16(q[1]);
    ab[p] = cv.u;
  }
  __syncthreads();
}

__device__ void wg_gemm_bf16(const __hip_bfloat16* abufp,
                             const __hip_bfloat16* W, int KCtot, int kc0,
                             int NT, const float* bias, float* dst,
                             int dstride, int dcol, bool relu, bool accum) {
  __syncthreads();
  const int lane = threadIdx.x & 31;
  const int wave = threadIdx.x >> 5;
  const v16bf* af = (const v16bf*)abufp;
  const v16bf* wf = (const v16bf*)W;
  v16bf a0 = af[(wave * 4 + 0) * 32 + lane];
  v16bf a1 = af[(wave * 4 + 1) * 32 + lane];
  v16bf a2 = af[(wave * 4 + 2) * 32 + lane];
  v16bf a3 = af[(wave * 4 + 3) * 32 + lane];
  const int row0 = wave * 16 + ((lane & 16) ? 8 : 0);
  for (int nt = 0; nt < NT; ++nt) {
    int col = dcol + nt * 16 + (lane & 15);
    const v16bf* wfn = wf + (nt * KCtot + kc0) * 32 + lane;
    __builtin_prefetch(wfn + KCtot * 32, 0, 0);     // next N-tile's B frags
    v8f acc;
    if (accum) {
#pragma unroll
      for (int v = 0; v < 8; ++v) acc[v] = dst[(row0 + v) * dstride + col];
    } else {
      float bv = bias[nt * 16 + (lane & 15)];
#pragma unroll
      for (int v = 0; v < 8; ++v) acc[v] = bv;
    }
    acc = __builtin_amdgcn_wmma_f32_16x16x32_bf16(false, a0, false, wfn[0],
                                                  (short)0, acc, false, false);
    acc = __builtin_amdgcn_wmma_f32_16x16x32_bf16(false, a1, false, wfn[32],
                                                  (short)0, acc, false, false);
    acc = __builtin_amdgcn_wmma_f32_16x16x32_bf16(false, a2, false, wfn[64],
                                                  (short)0, acc, false, false);
    acc = __builtin_amdgcn_wmma_f32_16x16x32_bf16(false, a3, false, wfn[96],
                                                  (short)0, acc, false, false);
#pragma unroll
    for (int v = 0; v < 8; ++v) {
      float val = acc[v];
      if (relu) val = fmaxf(val, 0.f);
      dst[(row0 + v) * dstride + col] = val;
    }
  }
}

__device__ void add_ln(float* xbuf, const float* ybuf, int ystride,
                       const float* g, const float* bta) {
  __syncthreads();
  int r = threadIdx.x;
  if (r < 128) {
    float* xr = xbuf + r * XS;
    const float* yr = ybuf + r * ystride;
    float s = 0.f, s2 = 0.f;
#pragma unroll 4
    for (int c = 0; c < 128; ++c) {
      float v = xr[c] + yr[c];
      s += v; s2 += v * v;
    }
    float m = s * (1.f / 128.f);
    float var = s2 * (1.f / 128.f) - m * m;
    float rs = rsqrtf(var + 1e-5f);
#pragma unroll 4
    for (int c = 0; c < 128; ++c) {
      float v = xr[c] + yr[c] - m;
      xr[c] = v * rs * g[c] + bta[c];
    }
  }
  __syncthreads();
}

__device__ void attn(float* hbuf) {
  float o[2][32];
#pragma unroll
  for (int p = 0; p < 2; ++p) {
    int pair = threadIdx.x + p * 256;        // 512 (token, head) pairs
    int i = pair & 63, h = pair >> 6;
    const float* r0 = hbuf + i * HS + h * 16;
    const float* r1 = hbuf + (64 + i) * HS + h * 16;
    float s00 = 0, s01 = 0, s10 = 0, s11 = 0;
#pragma unroll
    for (int d = 0; d < 16; ++d) {
      float q0 = r0[d], q1 = r1[d];
      float k0 = r0[128 + d], k1 = r1[128 + d];
      s00 = fmaf(q0, k0, s00); s01 = fmaf(q0, k1, s01);
      s10 = fmaf(q1, k0, s10); s11 = fmaf(q1, k1, s11);
    }
    const float sc = 0.25f;                  // 1/sqrt(16)
    s00 *= sc; s01 *= sc; s10 *= sc; s11 *= sc;
    float m0 = fmaxf(s00, s01), m1 = fmaxf(s10, s11);
    float e00 = __expf(s00 - m0), e01 = __expf(s01 - m0);
    float e10 = __expf(s10 - m1), e11 = __expf(s11 - m1);
    float i0 = 1.f / (e00 + e01), i1 = 1.f / (e10 + e11);
    float p00 = e00 * i0, p01 = e01 * i0, p10 = e10 * i1, p11 = e11 * i1;
#pragma unroll
    for (int d = 0; d < 16; ++d) {
      float v0 = r0[256 + d], v1 = r1[256 + d];
      o[p][d]      = p00 * v0 + p01 * v1;
      o[p][16 + d] = p10 * v0 + p11 * v1;
    }
  }
  __syncthreads();
#pragma unroll
  for (int p = 0; p < 2; ++p) {
    int pair = threadIdx.x + p * 256;
    int i = pair & 63, h = pair >> 6;
    float* r0 = hbuf + i * HS + h * 16;
    float* r1 = hbuf + (64 + i) * HS + h * 16;
#pragma unroll
    for (int d = 0; d < 16; ++d) { r0[d] = o[p][d]; r1[d] = o[p][16 + d]; }
  }
  __syncthreads();
}

__global__ void __launch_bounds__(256)
k3_xformer(const float* z2src, const __hip_bfloat16* wb,
           const float* qkv_b, const float* out_b,
           const float* ff1_b, const float* ff2_b,
           const float* ln1_g, const float* ln1_b,
           const float* ln2_g, const float* ln2_b,
           const float* dec_b, float* out_py) {
  extern __shared__ float sm[];
  float* xbuf = sm;                       // 128 x XS (f32 residual stream)
  float* hbuf = sm + 128 * XS;            // 128 x HS (qkv / o / ffn scratch)
  __hip_bfloat16* abuf = (__hip_bfloat16*)(sm + 128 * XS + 128 * HS); // 32KB
  const int tid = threadIdx.x;
  const size_t b0t = (size_t)blockIdx.x * 64;

  // load tile: row s*64+i <- z2[s, b0t+i, :]  (async global->LDS, b128 chunks)
#if USE_ASYNC
  for (int e = tid; e < 128 * 32; e += 256) {
    int row = e >> 5, c = (e & 31) * 4;
    int s = row >> 6, i = row & 63;
    size_t b = b0t + i;
    if (b < NB) {
      async_ld_b128(xbuf + row * XS + c, z2src + ((size_t)s * NB + b) * 128 + c);
    } else {
      float4 zz = {0.f, 0.f, 0.f, 0.f};
      *(float4*)(xbuf + row * XS + c) = zz;
    }
  }
  wait_async0();
#else
  for (int e = tid; e < 128 * 128; e += 256) {
    int row = e >> 7, c = e & 127;
    int s = row >> 6, i = row & 63;
    size_t b = b0t + i;
    float v = (b < NB) ? z2src[((size_t)s * NB + b) * 128 + c] : 0.f;
    xbuf[row * XS + c] = v;
  }
#endif
  mirror128(xbuf, XS, 0, abuf);                     // x -> bf16 A frags

  for (int l = 0; l < 5; ++l) {
    const __hip_bfloat16* wl = wb + l * 131072;
    // qkv: x @ (128x384) -> hbuf[:,0:384]
    wg_gemm_bf16(abuf, wl, 4, 0, 24, qkv_b + l * 384, hbuf, HS, 0, false, false);
    __syncthreads();
    attn(hbuf);                                     // o -> hbuf[:,0:128]
    mirror128(hbuf, HS, 0, abuf);                   // o -> A frags
    // out proj -> hbuf[:,128:256]
    wg_gemm_bf16(abuf, wl + 49152, 4, 0, 8, out_b + l * 128,
                 hbuf, HS, 128, false, false);
    add_ln(xbuf, hbuf + 128, HS, ln1_g + l * 128, ln1_b + l * 128);
    mirror128(xbuf, XS, 0, abuf);                   // x -> A frags
    // ff1 (relu) -> hbuf[:,0:256]
    wg_gemm_bf16(abuf, wl + 65536, 4, 0, 16, ff1_b + l * 256,
                 hbuf, HS, 0, true, false);
    mirror128(hbuf, HS, 0, abuf);                   // h[:,0:128] -> A frags
    // ff2 half 1 -> hbuf[:,256:384]
    wg_gemm_bf16(abuf, wl + 98304, 8, 0, 8, ff2_b + l * 128,
                 hbuf, HS, 256, false, false);
    mirror128(hbuf, HS, 128, abuf);                 // h[:,128:256] -> A frags
    // ff2 half 2 (accumulate)
    wg_gemm_bf16(abuf, wl + 98304, 8, 4, 8, ff2_b + l * 128,
                 hbuf, HS, 256, false, true);
    add_ln(xbuf, hbuf + 256, HS, ln2_g + l * 128, ln2_b + l * 128);
    mirror128(xbuf, XS, 0, abuf);                   // x -> A frags (next layer)
  }

  // decoder: x @ (128x64) -> hbuf[:,0:64], then mean over seq slots
  wg_gemm_bf16(abuf, wb + 655360, 4, 0, 4, dec_b, hbuf, HS, 0, false, false);
  __syncthreads();
  for (int e = tid; e < 64 * 64; e += 256) {
    int i = e & 63, c = e >> 6;
    size_t b = b0t + i;
    if (b < NB) {
      float v = 0.5f * (hbuf[i * HS + c] + hbuf[(64 + i) * HS + c]);
      out_py[(size_t)c * NB + b] = v;
    }
  }
}

// ---------------------------------------------------------------------------
extern "C" void kernel_launch(void* const* d_in, const int* in_sizes, int n_in,
                              void* d_out, int out_size, void* d_ws, size_t ws_size,
                              hipStream_t stream) {
  const float* t     = (const float*)d_in[0];
  const float* V     = (const float*)d_in[2];   // shooting_vars
  const float* A     = (const float*)d_in[3];
  const float* qkv_w = (const float*)d_in[4];
  const float* qkv_b = (const float*)d_in[5];
  const float* out_w = (const float*)d_in[6];
  const float* out_b = (const float*)d_in[7];
  const float* ff1_w = (const float*)d_in[8];
  const float* ff1_b = (const float*)d_in[9];
  const float* ff2_w = (const float*)d_in[10];
  const float* ff2_b = (const float*)d_in[11];
  const float* ln1_g = (const float*)d_in[12];
  const float* ln1_b = (const float*)d_in[13];
  const float* ln2_g = (const float*)d_in[14];
  const float* ln2_b = (const float*)d_in[15];
  const float* dec_w = (const float*)d_in[16];
  const float* dec_b = (const float*)d_in[17];

  float* out     = (float*)d_out;
  float* out_py  = out;                    // (64, 131073)
  float* out_z2  = out + 8388672;          // (2, 131073, 128)
  float* out_z20 = out_z2 + 33554688;      // (131073, 128)
  float* out_z21 = out_z20 + 16777344;     // (131073, 128)

  float* wsf = (float*)d_ws;
  __hip_bfloat16* wb = (__hip_bfloat16*)((char*)d_ws + WB_BYTE_OFF);

  // K0: one-time weight swizzle to bf16 WMMA fragments (663552 elements)
  k0_swz<<<2592, 256, 0, stream>>>(qkv_w, out_w, ff1_w, ff2_w, dec_w, wb);
  // K1: RK4 one-step matrix, power table, checkpoints (1 WG, f32 WMMA)
  k1_mats<<<1, 256, 3 * 129 * DLS * 4, stream>>>(A, t, V, wsf);
  // K2: 128 parallel 16-step segments + z2 scatter (f32 WMMA + async movers)
  k2_shoot<<<128, 256, (2 * 129 * DLS + 128 * DLS) * 4, stream>>>(
      wsf, out_z2, out_z20, out_z21);
  // K3: fused transformer + decoder (bf16 WMMA), 2049 tiles of 64 tokens
  k3_xformer<<<2049, 256, (128 * XS + 128 * HS) * 4 + 128 * 128 * 2, stream>>>(
      out_z2, wb, qkv_b, out_b, ff1_b, ff2_b,
      ln1_g, ln1_b, ln2_g, ln2_b, dec_b, out_py);
}
